// Attention_39711267619067
// MI455X (gfx1250) — matrix operand order repaired and checked by reading, not compiled, not run on hardware
//
#include <hip/hip_runtime.h>

// ---------------------------------------------------------------------------
// MLA attention forward for MI455X (gfx1250, wave32, WMMA bf16,
// async global->LDS staging, double-buffered GEMM).
// ---------------------------------------------------------------------------

typedef unsigned short u16;
typedef __attribute__((ext_vector_type(16))) __bf16 v16bf;
typedef __attribute__((ext_vector_type(8)))  float  v8f;

union Frag { v16bf v; uint4 q[2]; };
union H8   { uint4 q; u16 h[8]; };

__device__ __forceinline__ u16 f2bf(float f) {
  unsigned u = __float_as_uint(f);
  u += 0x7fffu + ((u >> 16) & 1u);           // round-to-nearest-even
  return (u16)(u >> 16);
}

// LDS byte address (within workgroup allocation) of a __shared__ object.
__device__ __forceinline__ unsigned lds_addr(const void* p) {
  return (unsigned)(unsigned long long)
      (const __attribute__((address_space(3))) void*)p;
}

// Async global->LDS copy, 16 bytes per lane. ASYNCcnt-tracked (ISA 15.18.3).
// GVS addressing: mem = SADDR(64b sgpr) + VADDR(32b vgpr byte offset).
__device__ __forceinline__ void async_b128(unsigned lds_off, const void* base,
                                           unsigned byte_off) {
  asm volatile("global_load_async_to_lds_b128 %0, %1, %2"
               :: "v"(lds_off), "v"(byte_off),
                  "s"((unsigned long long)(size_t)base)
               : "memory");
}
__device__ __forceinline__ void wait_async0() {
  asm volatile("s_wait_asynccnt 0x0" ::: "memory");
}

#define SEQ   2048
#define HEADS 16
#define QKD_  192
#define ATT_SCALE 0.07216878364870322f       // 192^-0.5

// ---------------------------------------------------------------------------
// fp32 -> bf16 cast (grid-stride)
// ---------------------------------------------------------------------------
__global__ void k_cast_bf16(const float* __restrict__ in, u16* __restrict__ out, int n) {
  for (int i = blockIdx.x * blockDim.x + threadIdx.x; i < n; i += gridDim.x * blockDim.x)
    out[i] = f2bf(in[i]);
}

// ---------------------------------------------------------------------------
// fp32 [K,N] -> bf16 [N,K] transposed cast (32x32 tiles through LDS).
// ---------------------------------------------------------------------------
__global__ __launch_bounds__(256) void k_cast_t(const float* __restrict__ in,
    u16* __restrict__ out, int K, int N) {
  __shared__ float t[32][33];
  const int tx = threadIdx.x & 31, ty = threadIdx.x >> 5;   // 8 rows / pass
  const int kBase = blockIdx.y * 32, nBase = blockIdx.x * 32;
  #pragma unroll
  for (int r = 0; r < 4; ++r)
    t[ty + r * 8][tx] = in[(size_t)(kBase + ty + r * 8) * N + nBase + tx];
  __syncthreads();
  #pragma unroll
  for (int r = 0; r < 4; ++r)
    out[(size_t)(nBase + ty + r * 8) * K + kBase + tx] = f2bf(t[tx][ty + r * 8]);
}

// ---------------------------------------------------------------------------
// bf16 WMMA GEMM: C[M,N] = A[M,K] x Bt[N,K]^T ; C f32.
// 256 threads = 8 waves; block tile 128x64; wave tile 32x32; K step 64;
// double-buffered LDS stages filled with global_load_async_to_lds_b128.
// ---------------------------------------------------------------------------
__global__ __launch_bounds__(256) void k_gemm_bf16(
    const u16* __restrict__ A, const u16* __restrict__ Bt, float* __restrict__ C,
    int M, int N, int K) {
  __shared__ u16 As[2][128 * 64];   // [stage][row][k]  2 x 16 KB
  __shared__ u16 Bs[2][64 * 64];    // [stage][n][k]    2 x  8 KB
  const int tid  = threadIdx.x;
  const int wave = tid >> 5, lane = tid & 31;
  const int wr = wave >> 1, wc = wave & 1;
  const int l15 = lane & 15, hi = lane >> 4;
  const size_t rowBase = (size_t)blockIdx.y * 128;
  const size_t colBase = (size_t)blockIdx.x * 64;

  // Per-thread staging coordinates (contiguous 16B chunks).
  const int ar = (tid * 32) >> 6, ac = (tid * 32) & 63;   // A: 32 halves/thread
  const int br = (tid * 16) >> 6, bc = (tid * 16) & 63;   // B: 16 halves/thread
  const unsigned aRow2 = (unsigned)(((rowBase + ar) * (size_t)K + ac) * 2);
  const unsigned bRow2 = (unsigned)(((colBase + br) * (size_t)K + bc) * 2);
  const unsigned lA0 = lds_addr(&As[0][ar * 64 + ac]);
  const unsigned lB0 = lds_addr(&Bs[0][br * 64 + bc]);
  const unsigned sStrideA = 128 * 64 * 2, sStrideB = 64 * 64 * 2;

  // prologue: stage k-slice 0 into buffer 0
  #pragma unroll
  for (int e = 0; e < 4; ++e) async_b128(lA0 + e * 16, A,  aRow2 + e * 16);
  #pragma unroll
  for (int e = 0; e < 2; ++e) async_b128(lB0 + e * 16, Bt, bRow2 + e * 16);

  v8f acc[2][2] = {};
  const int nIter = K >> 6;
  for (int i = 0; i < nIter; ++i) {
    wait_async0();
    __syncthreads();
    if (i + 1 < nIter) {                       // prefetch next slice, other buf
      const unsigned s = (i + 1) & 1;
      const unsigned kOfs = (unsigned)((i + 1) << 6) * 2;
      #pragma unroll
      for (int e = 0; e < 4; ++e)
        async_b128(lA0 + s * sStrideA + e * 16, A,  aRow2 + kOfs + e * 16);
      #pragma unroll
      for (int e = 0; e < 2; ++e)
        async_b128(lB0 + s * sStrideB + e * 16, Bt, bRow2 + kOfs + e * 16);
    }
    const u16* as = &As[i & 1][0];
    const u16* bs = &Bs[i & 1][0];
    #pragma unroll
    for (int kk = 0; kk < 64; kk += 32) {
      Frag a[2], b[2];
      #pragma unroll
      for (int mt = 0; mt < 2; ++mt) {
        int row = wr * 32 + mt * 16 + l15;
        a[mt].q[0] = *(const uint4*)&as[row * 64 + kk + hi * 8];
        a[mt].q[1] = *(const uint4*)&as[row * 64 + kk + hi * 8 + 16];
      }
      #pragma unroll
      for (int nt = 0; nt < 2; ++nt) {
        int n = wc * 32 + nt * 16 + l15;
        b[nt].q[0] = *(const uint4*)&bs[n * 64 + kk + hi * 8];
        b[nt].q[1] = *(const uint4*)&bs[n * 64 + kk + hi * 8 + 16];
      }
      #pragma unroll
      for (int mt = 0; mt < 2; ++mt)
        #pragma unroll
        for (int nt = 0; nt < 2; ++nt)
          acc[mt][nt] = __builtin_amdgcn_wmma_f32_16x16x32_bf16(
              false, a[mt].v, false, b[nt].v, (short)0, acc[mt][nt], false, false);
    }
  }
  #pragma unroll
  for (int mt = 0; mt < 2; ++mt)
    #pragma unroll
    for (int nt = 0; nt < 2; ++nt)
      #pragma unroll
      for (int r = 0; r < 8; ++r) {
        size_t row = rowBase + wr * 32 + mt * 16 + hi * 8 + r;
        size_t col = colBase + wc * 32 + nt * 16 + l15;
        C[row * N + col] = acc[mt][nt][r];
      }
}

// ---------------------------------------------------------------------------
// RMS norm over rows -> bf16
// ---------------------------------------------------------------------------
__global__ __launch_bounds__(256) void k_rmsnorm(const float* __restrict__ in,
    const float* __restrict__ w, u16* __restrict__ out, int cols) {
  __shared__ float red[256];
  const size_t row = blockIdx.x;
  const float* x = in + row * cols;
  float s = 0.f;
  for (int c = threadIdx.x; c < cols; c += 256) { float v = x[c]; s += v * v; }
  red[threadIdx.x] = s;
  __syncthreads();
  for (int o = 128; o > 0; o >>= 1) {
    if (threadIdx.x < o) red[threadIdx.x] += red[threadIdx.x + o];
    __syncthreads();
  }
  float rstd = rsqrtf(red[0] / (float)cols + 1e-6f);
  u16* y = out + row * cols;
  for (int c = threadIdx.x; c < cols; c += 256) y[c] = f2bf(x[c] * rstd * w[c]);
}

// ---------------------------------------------------------------------------
// kv_a epilogue: rmsnorm first 512 cols -> kvn bf16; RoPE last 64 -> kpe bf16.
// ---------------------------------------------------------------------------
__global__ __launch_bounds__(256) void k_kva_proc(const float* __restrict__ kva,
    const float* __restrict__ w, const float* __restrict__ freqs,
    u16* __restrict__ kvn, u16* __restrict__ kpe) {
  __shared__ float red[256];
  const int row = blockIdx.x;
  const int s = row & (SEQ - 1);
  const float* x = kva + (size_t)row * 576;
  float acc = 0.f;
  #pragma unroll
  for (int i = 0; i < 2; ++i) { float v = x[threadIdx.x + 256 * i]; acc += v * v; }
  red[threadIdx.x] = acc;
  __syncthreads();
  for (int o = 128; o > 0; o >>= 1) {
    if (threadIdx.x < o) red[threadIdx.x] += red[threadIdx.x + o];
    __syncthreads();
  }
  float rstd = rsqrtf(red[0] / 512.f + 1e-6f);
  #pragma unroll
  for (int i = 0; i < 2; ++i) {
    int c = threadIdx.x + 256 * i;
    kvn[(size_t)row * 512 + c] = f2bf(x[c] * rstd * w[c]);
  }
  if (threadIdx.x < 32) {
    int p = threadIdx.x;
    float e = x[512 + 2 * p], o = x[512 + 2 * p + 1];
    float cs = freqs[((size_t)s * 32 + p) * 2 + 0];
    float sn = freqs[((size_t)s * 32 + p) * 2 + 1];
    kpe[(size_t)row * 64 + 2 * p]     = f2bf(e * cs - o * sn);
    kpe[(size_t)row * 64 + 2 * p + 1] = f2bf(e * sn + o * cs);
  }
}

// ---------------------------------------------------------------------------
// q epilogue: RoPE last 64 of each head, relayout to [B,H,S,192] bf16.
// ---------------------------------------------------------------------------
__global__ __launch_bounds__(256) void k_qbuild(const float* __restrict__ q,
    const float* __restrict__ freqs, u16* __restrict__ q_r) {
  const int row = blockIdx.x;                 // b*S + s
  const int b = row >> 11, s = row & (SEQ - 1);
  const float* in = q + (size_t)row * (HEADS * QKD_);
  for (int pos = threadIdx.x; pos < HEADS * 96; pos += 256) {
    int h = pos / 96, t = pos - h * 96;
    const float* hin = in + h * QKD_;
    u16* hout = q_r + ((size_t)(b * HEADS + h) * SEQ + s) * QKD_;
    if (t < 64) {
      hout[2 * t]     = f2bf(hin[2 * t]);
      hout[2 * t + 1] = f2bf(hin[2 * t + 1]);
    } else {
      int p = t - 64;
      float e = hin[128 + 2 * p], o = hin[128 + 2 * p + 1];
      float cs = freqs[((size_t)s * 32 + p) * 2], sn = freqs[((size_t)s * 32 + p) * 2 + 1];
      hout[128 + 2 * p]     = f2bf(e * cs - o * sn);
      hout[128 + 2 * p + 1] = f2bf(e * sn + o * cs);
    }
  }
}

// ---------------------------------------------------------------------------
// k/v epilogue: split kv -> k_nope/v, append broadcast k_pe; [B,H,S,·] bf16.
// ---------------------------------------------------------------------------
__global__ __launch_bounds__(256) void k_kvbuild(const float* __restrict__ kv,
    const u16* __restrict__ kpe, u16* __restrict__ k_r, u16* __restrict__ v_r) {
  const int row = blockIdx.x;
  const int b = row >> 11, s = row & (SEQ - 1);
  const float* in = kv + (size_t)row * 4096;
  for (int pos = threadIdx.x; pos < HEADS * 256; pos += 256) {
    int h = pos >> 8, d = pos & 255;
    size_t hs = (size_t)(b * HEADS + h) * SEQ + s;
    float v = in[h * 256 + d];
    if (d < 128) k_r[hs * QKD_ + d] = f2bf(v);
    else         v_r[hs * 128 + (d - 128)] = f2bf(v);
  }
  for (int pos = threadIdx.x; pos < HEADS * 64; pos += 256) {
    int h = pos >> 6, d = pos & 63;
    size_t hs = (size_t)(b * HEADS + h) * SEQ + s;
    k_r[hs * QKD_ + 128 + d] = kpe[(size_t)row * 64 + d];
  }
}

// ---------------------------------------------------------------------------
// Flash attention: 4 waves, 64-row Q block, causal, online softmax.
// K tile staged with async global->LDS (overlaps the V transpose).
// ---------------------------------------------------------------------------
__global__ __launch_bounds__(128) void k_attn(const u16* __restrict__ q_r,
    const u16* __restrict__ k_r, const u16* __restrict__ v_r,
    u16* __restrict__ o_bf) {
  __shared__ u16 Ks[64 * QKD_];     // K tile row-major   24 KB
  __shared__ u16 Vts[128 * 64];     // V tile transposed  16 KB
  __shared__ u16 Ps[4 * 16 * 64];   // per-wave P strips   8 KB
  const int qb = blockIdx.x, h = blockIdx.y, b = blockIdx.z;
  const int tid = threadIdx.x;
  const int w = tid >> 5, lane = tid & 31;
  const int l15 = lane & 15, hi = lane >> 4;
  const size_t bh = (size_t)(b * HEADS + h) * SEQ;

  Frag qf[6];                                   // Q fragments resident (16x192)
  {
    const u16* qrow = q_r + (bh + qb * 64 + w * 16 + l15) * QKD_;
    #pragma unroll
    for (int ks = 0; ks < 6; ++ks) {
      int kb = ks * 32 + hi * 8;
      qf[ks].q[0] = *(const uint4*)(qrow + kb);
      qf[ks].q[1] = *(const uint4*)(qrow + kb + 16);
    }
  }
  v8f oa[8] = {};
  float m_run[8], l_run[8];
  #pragma unroll
  for (int r = 0; r < 8; ++r) { m_run[r] = -1e30f; l_run[r] = 0.f; }

  const unsigned ksBase = lds_addr(&Ks[0]);

  for (int j = 0; j <= qb; ++j) {
    __syncthreads();                            // prior-iter LDS reads done
    { // stage K tile via async copies (12 x 16B per thread)
      const u16* kg = k_r + (bh + j * 64) * QKD_;
      #pragma unroll
      for (int c = 0; c < 12; ++c) {
        int i = tid + c * 128;
        async_b128(ksBase + (unsigned)i * 16, kg, (unsigned)i * 16);
      }
    }
    { // stage V tile transposed (VGPR path, overlaps async K copy)
      const u16* vg = v_r + (bh + j * 64) * 128;
      for (int i = tid; i < 1024; i += 128) {
        int linear = i * 8;
        int jr = linear >> 7, d0 = linear & 127;
        H8 hv; hv.q = *(const uint4*)(vg + jr * 128 + d0);
        #pragma unroll
        for (int e = 0; e < 8; ++e) Vts[(d0 + e) * 64 + jr] = hv.h[e];
      }
    }
    wait_async0();
    __syncthreads();

    v8f sc[4] = {};                             // 16x64 score strip
    #pragma unroll
    for (int ks = 0; ks < 6; ++ks) {
      #pragma unroll
      for (int ct = 0; ct < 4; ++ct) {
        Frag bf;
        int krow = ct * 16 + l15;
        int kb2 = ks * 32 + hi * 8;
        bf.q[0] = *(const uint4*)&Ks[krow * QKD_ + kb2];
        bf.q[1] = *(const uint4*)&Ks[krow * QKD_ + kb2 + 16];
        sc[ct] = __builtin_amdgcn_wmma_f32_16x16x32_bf16(
            false, qf[ks].v, false, bf.v, (short)0, sc[ct], false, false);
      }
    }
    // scale + causal mask
    const int qrg  = qb * 64 + w * 16 + hi * 8;
    const int kcg0 = j * 64 + l15;
    #pragma unroll
    for (int ct = 0; ct < 4; ++ct)
      #pragma unroll
      for (int r = 0; r < 8; ++r) {
        float v = sc[ct][r] * ATT_SCALE;
        if (kcg0 + ct * 16 > qrg + r) v = -1e30f;
        sc[ct][r] = v;
      }
    // online softmax (row stats replicated across each 16-lane half)
    float mNew[8], corr[8];
    #pragma unroll
    for (int r = 0; r < 8; ++r) {
      float m = fmaxf(fmaxf(sc[0][r], sc[1][r]), fmaxf(sc[2][r], sc[3][r]));
      #pragma unroll
      for (int off = 1; off < 16; off <<= 1)
        m = fmaxf(m, __shfl_xor(m, off, 32));
      mNew[r] = fmaxf(m_run[r], m);
      corr[r] = __expf(m_run[r] - mNew[r]);
    }
    #pragma unroll
    for (int ct = 0; ct < 4; ++ct)
      #pragma unroll
      for (int r = 0; r < 8; ++r)
        sc[ct][r] = __expf(sc[ct][r] - mNew[r]);
    #pragma unroll
    for (int r = 0; r < 8; ++r) {
      float s = sc[0][r] + sc[1][r] + sc[2][r] + sc[3][r];
      #pragma unroll
      for (int off = 1; off < 16; off <<= 1)
        s += __shfl_xor(s, off, 32);
      l_run[r] = l_run[r] * corr[r] + s;
      m_run[r] = mNew[r];
    }
    #pragma unroll
    for (int nt = 0; nt < 8; ++nt)
      #pragma unroll
      for (int r = 0; r < 8; ++r)
        oa[nt][r] *= corr[r];
    // C-layout f32 -> A-layout bf16 via per-wave LDS strip
    #pragma unroll
    for (int ct = 0; ct < 4; ++ct)
      #pragma unroll
      for (int r = 0; r < 8; ++r)
        Ps[(w * 16 + hi * 8 + r) * 64 + ct * 16 + l15] = f2bf(sc[ct][r]);
    // P x V  (16x64 . 64x128)
    #pragma unroll
    for (int kc = 0; kc < 2; ++kc) {
      Frag pa;
      int kb = kc * 32 + hi * 8;
      pa.q[0] = *(const uint4*)&Ps[(w * 16 + l15) * 64 + kb];
      pa.q[1] = *(const uint4*)&Ps[(w * 16 + l15) * 64 + kb + 16];
      #pragma unroll
      for (int nt = 0; nt < 8; ++nt) {
        Frag vb;
        int d = nt * 16 + l15;
        int kb2 = kc * 32 + hi * 8;
        vb.q[0] = *(const uint4*)&Vts[d * 64 + kb2];
        vb.q[1] = *(const uint4*)&Vts[d * 64 + kb2 + 16];
        oa[nt] = __builtin_amdgcn_wmma_f32_16x16x32_bf16(
            false, pa.v, false, vb.v, (short)0, oa[nt], false, false);
      }
    }
  }
  // normalize + write [B,S,H*128] bf16
  const int qrow = qb * 64 + w * 16 + hi * 8;
  #pragma unroll
  for (int nt = 0; nt < 8; ++nt)
    #pragma unroll
    for (int r = 0; r < 8; ++r) {
      size_t idx = ((size_t)b * SEQ + (qrow + r)) * 2048 + h * 128 + nt * 16 + l15;
      o_bf[idx] = f2bf(oa[nt][r] / l_run[r]);
    }
}

// ---------------------------------------------------------------------------
// Host-side orchestration
// ---------------------------------------------------------------------------
extern "C" void kernel_launch(void* const* d_in, const int* in_sizes, int n_in,
                              void* d_out, int out_size, void* d_ws, size_t ws_size,
                              hipStream_t stream) {
  (void)in_sizes; (void)n_in; (void)out_size; (void)ws_size;
  const float* x       = (const float*)d_in[0];
  const float* freqs   = (const float*)d_in[1];
  const float* wq_a    = (const float*)d_in[2];
  const float* q_norm  = (const float*)d_in[3];
  const float* wq_b    = (const float*)d_in[4];
  const float* wkv_a   = (const float*)d_in[5];
  const float* kv_norm = (const float*)d_in[6];
  const float* wkv_b   = (const float*)d_in[7];
  const float* wo      = (const float*)d_in[8];
  float* out = (float*)d_out;

  char* ws = (char*)d_ws;
  size_t off = 0;
  auto alloc = [&](size_t bytes) {
    char* p = ws + off; off += (bytes + 255) & ~(size_t)255; return (void*)p;
  };
  const int M = 2 * SEQ;   // 4096 rows
  u16*   x_bf    = (u16*)  alloc((size_t)M * 2048 * 2);
  u16*   wqa_t   = (u16*)  alloc((size_t)2048 * 1536 * 2);   // [N,K] bf16
  u16*   wqb_t   = (u16*)  alloc((size_t)1536 * 3072 * 2);
  u16*   wkva_t  = (u16*)  alloc((size_t)2048 * 576 * 2);
  u16*   wkvb_t  = (u16*)  alloc((size_t)512 * 4096 * 2);
  u16*   wo_t    = (u16*)  alloc((size_t)2048 * 2048 * 2);
  float* bufA    = (float*)alloc((size_t)M * 1536 * 4);      // qa, then kv_a
  float* bufB    = (float*)alloc((size_t)M * 4096 * 4);      // q, then kv
  u16*   nrm_bf  = (u16*)  alloc((size_t)M * 1536 * 2);      // qn, then kvn
  u16*   kpe_bf  = (u16*)  alloc((size_t)M * 64 * 2);
  u16*   q_r     = (u16*)  alloc((size_t)M * HEADS * QKD_ * 2);
  u16*   k_r     = (u16*)  alloc((size_t)M * HEADS * QKD_ * 2);
  u16*   v_r     = (u16*)  alloc((size_t)M * HEADS * 128 * 2);
  u16*   o_bf    = (u16*)  alloc((size_t)M * 2048 * 2);

  k_cast_bf16<<<1024, 256, 0, stream>>>(x, x_bf, M * 2048);

  auto castT = [&](const float* src, u16* dst, int K, int N) {
    dim3 grid(N / 32, K / 32);
    k_cast_t<<<grid, 256, 0, stream>>>(src, dst, K, N);
  };
  castT(wq_a,  wqa_t,  2048, 1536);
  castT(wq_b,  wqb_t,  1536, 3072);
  castT(wkv_a, wkva_t, 2048, 576);
  castT(wkv_b, wkvb_t, 512,  4096);
  castT(wo,    wo_t,   2048, 2048);

  auto gemm = [&](const u16* A, const u16* Bt, float* C, int m, int n, int k) {
    dim3 grid(n / 64, m / 128);
    k_gemm_bf16<<<grid, 256, 0, stream>>>(A, Bt, C, m, n, k);
  };

  gemm(x_bf, wqa_t, bufA, M, 1536, 2048);                    // x @ wq_a
  k_rmsnorm<<<M, 256, 0, stream>>>(bufA, q_norm, nrm_bf, 1536);
  gemm(nrm_bf, wqb_t, bufB, M, 3072, 1536);                  // qn @ wq_b
  k_qbuild<<<M, 256, 0, stream>>>(bufB, freqs, q_r);
  gemm(x_bf, wkva_t, bufA, M, 576, 2048);                    // x @ wkv_a
  k_kva_proc<<<M, 256, 0, stream>>>(bufA, kv_norm, freqs, nrm_bf, kpe_bf);
  gemm(nrm_bf, wkvb_t, bufB, M, 4096, 512);                  // kvn @ wkv_b
  k_kvbuild<<<M, 256, 0, stream>>>(bufB, kpe_bf, k_r, v_r);
  dim3 agrid(SEQ / 64, HEADS, 2);
  k_attn<<<agrid, 128, 0, stream>>>(q_r, k_r, v_r, o_bf);    // flash attention
  gemm(o_bf, wo_t, out, M, 2048, 2048);                      // @ wo -> f32 out
}
